// AttentionSeqToMasked_9818295239319
// MI455X (gfx1250) — matrix operand order repaired and checked by reading, not compiled, not run on hardware
//
#include <hip/hip_runtime.h>
#include <hip/hip_bf16.h>

// ---------------------------------------------------------------------------
// AttentionSeqToMasked on MI455X (gfx1250, wave32, WMMA)
//   phase 1: proj16<VT>  -- Q/K/V projection GEMMs, f32 -> f16, WMMA f16/f32
//   phase 2: flash_attn  -- flash attention, WMMA f16/f32, online softmax
// ---------------------------------------------------------------------------

typedef __attribute__((ext_vector_type(16))) _Float16 v16h;
typedef __attribute__((ext_vector_type(8)))  _Float16 v8h;
typedef __attribute__((ext_vector_type(8)))  float    v8f;

#define B_    4
#define SQ_   2048
#define SK_   2048
#define DPRE  1024
#define H_    16
#define DQK   64
#define DV    64
#define NOUT  (H_ * DQK)   // 1024

static __device__ __forceinline__ v8f wmma_f16(v16h a, v16h b, v8f c) {
  // D = A(16x32 f16) * B(32x16 f16) + C(16x16 f32)
  return __builtin_amdgcn_wmma_f32_16x16x32_f16(
      /*neg_a=*/false, a, /*neg_b=*/false, b,
      /*c_mod=*/(short)0, c, /*reuse_a=*/false, /*reuse_b=*/false);
}

// ---------------------------------------------------------------------------
// Projection GEMM: dst = x @ w + bias, head-split fused into the store.
//   x: [8192, 1024] f32,  w: [1024, 1024] f32, bias: [1024] f32
//   VT == 0: dst[b][h][s][d]  (Q, K)
//   VT == 1: dst[b][h][d][s]  (V transposed, so PV B-fragments are contiguous)
// Block: 256 thr = 8 waves as 4(M) x 2(N); tile 64(M) x 128(N) x 64(K-stage).
// ---------------------------------------------------------------------------
template <int VT>
__global__ __launch_bounds__(256) void proj16(
    const float* __restrict__ x, const float* __restrict__ w,
    const float* __restrict__ bias, _Float16* __restrict__ dst) {
  // row stride 72 halves = 144 B (16B-aligned rows, +pad vs. 64 banks)
  __shared__ _Float16 a_s[64 * 72];    // [row][kk]
  __shared__ _Float16 b_t[128 * 72];   // [n][kk] (transposed: contiguous frags)

  const int tid  = threadIdx.x;
  const int lane = tid & 31;
  const int wid  = tid >> 5;
  const int wm   = wid >> 1;                 // 0..3 : 16-row slice
  const int wn   = wid & 1;                  // 0..1 : 64-col slice
  const int m_base = blockIdx.x * 64;
  const int n_base = blockIdx.y * 128;

  v8f acc[4] = {};                           // 4 x (16x16 f32) along N

  const int arow  = wm * 16 + (lane & 15);
  const int akoff = (lane >> 4) * 16;        // A frag: hi half-wave +16 in K
  const int bkoff = (lane >> 4) * 16;        // B frag: hi half-wave +16 in K
  const int n16   = lane & 15;

  for (int k0 = 0; k0 < DPRE; k0 += 64) {
    // ---- stage A tile 64x64 (f32 -> f16), vectorized ds_store_b128 ----------
#pragma unroll
    for (int i = 0; i < 2; ++i) {
      int idx = tid * 16 + i * 8;            // 0..4095 = r*64 + c, c multiple of 8
      int r = idx >> 6, c = idx & 63;
      const float* src = x + (size_t)(m_base + r) * DPRE + k0 + c;
      v8h tmp;
#pragma unroll
      for (int q = 0; q < 8; ++q) tmp[q] = (_Float16)src[q];
      *(v8h*)&a_s[r * 72 + c] = tmp;
    }
    // ---- stage B tile 64x128 transposed into [n][kk] ------------------------
    // assignment a = i*256+tid: n = a & 127 (coalesced reads), kkb = a >> 7
#pragma unroll
    for (int i = 0; i < 4; ++i) {
      int a = i * 256 + tid;                 // 0..1023
      int n = a & 127, kkb = a >> 7;         // 8-wide kk block
      const float* src = w + (size_t)(k0 + kkb * 8) * NOUT + n_base + n;
      v8h tmp;
#pragma unroll
      for (int q = 0; q < 8; ++q) tmp[q] = (_Float16)src[(size_t)q * NOUT];
      *(v8h*)&b_t[n * 72 + kkb * 8] = tmp;
    }
    __syncthreads();

#pragma unroll
    for (int kc = 0; kc < 2; ++kc) {
      // A fragment (16x32): halves 0-7 -> K=akoff+0..7, 8-15 -> K=akoff+8..15
      const _Float16* ap = &a_s[arow * 72 + kc * 32 + akoff];
      v8h alo = *(const v8h*)ap;
      v8h ahi = *(const v8h*)(ap + 8);
      v16h afrag;
#pragma unroll
      for (int hh = 0; hh < 8; ++hh) { afrag[hh] = alo[hh]; afrag[hh + 8] = ahi[hh]; }

#pragma unroll
      for (int j = 0; j < 4; ++j) {
        int n = wn * 64 + j * 16 + n16;
        // B fragment (32x16): half h -> K = bkoff + h (contiguous in b_t row)
        const _Float16* bp = &b_t[n * 72 + kc * 32 + bkoff];
        v8h blo = *(const v8h*)bp;
        v8h bhi = *(const v8h*)(bp + 8);
        v16h bfrag;
#pragma unroll
        for (int hh = 0; hh < 8; ++hh) { bfrag[hh] = blo[hh]; bfrag[hh + 8] = bhi[hh]; }
        acc[j] = wmma_f16(afrag, bfrag, acc[j]);
      }
    }
    __syncthreads();
  }

  // ---- epilogue: bias + head split (branch-free; VT is a template arg) -----
  // C layout: lane group has column n16; VGPR slot r -> row r (+8 hi half-wave)
  const int rowoff = (lane >> 4) * 8;
  const int gm0 = m_base + wm * 16 + rowoff;
  const int bb  = gm0 >> 11;                  // SQ=2048; 64-row tiles never cross b
  const int ss0 = gm0 & (SQ_ - 1);
#pragma unroll
  for (int j = 0; j < 4; ++j) {
    int gn = n_base + wn * 64 + j * 16 + n16;
    float bv = bias[gn];
    int hh = gn >> 6, d = gn & 63;
    _Float16* dp = VT
        ? dst + (((size_t)bb * H_ + hh) * DQK + d) * SK_ + ss0   // [b][h][d][s]
        : dst + (((size_t)bb * H_ + hh) * SQ_ + ss0) * DQK + d;  // [b][h][s][d]
    const size_t stride = VT ? 1u : (size_t)DQK;                 // step per row
#pragma unroll
    for (int r = 0; r < 8; ++r)
      dp[(size_t)r * stride] = (_Float16)(acc[j][r] + bv);
  }
}

// ---------------------------------------------------------------------------
// Flash attention. Grid (SQ/64, H, B), block 128 = 4 waves; each wave owns a
// 16-row Q tile and streams KV in tiles of 32.
// ---------------------------------------------------------------------------
__global__ __launch_bounds__(128) void flash_attn(
    const _Float16* __restrict__ Q,   // [B,H,SQ,64]
    const _Float16* __restrict__ K,   // [B,H,SK,64]
    const _Float16* __restrict__ Vt,  // [B,H,64,SK]
    const unsigned char* __restrict__ mask,  // [B,SK] (bool)
    float* __restrict__ out) {        // [B,SQ,H*64]
  __shared__ _Float16 p_s[4 * 16 * 40];      // per-wave P staging (C -> A relayout)

  const int lane = threadIdx.x & 31;
  const int wave = threadIdx.x >> 5;
  const int b = blockIdx.z, h = blockIdx.y;
  const int q0 = blockIdx.x * 64 + wave * 16;
  const size_t bh = (size_t)b * H_ + h;
  const _Float16* Qb = Q + bh * SQ_ * DQK;
  const _Float16* Kb = K + bh * SK_ * DQK;
  const _Float16* Vb = Vt + bh * DQK * SK_;
  const unsigned char* mb = mask + (size_t)b * SK_;
  _Float16* pw = p_s + wave * 16 * 40;

  const int arow   = lane & 15;
  const int akoff  = (lane >> 4) * 16;
  const int n16    = lane & 15;
  const int bkoff  = (lane >> 4) * 16;
  const int rowoff = (lane >> 4) * 8;
  const float scale = 0.125f;                // 1/sqrt(64)

  // Q fragments: two 16x32 k-chunks of the 16x64 Q tile
  v16h qf[2];
#pragma unroll
  for (int kc = 0; kc < 2; ++kc) {
    const _Float16* p = Qb + (size_t)(q0 + arow) * DQK + kc * 32 + akoff;
    v8h lo = *(const v8h*)p;
    v8h hi = *(const v8h*)(p + 8);
#pragma unroll
    for (int hh = 0; hh < 8; ++hh) { qf[kc][hh] = lo[hh]; qf[kc][hh + 8] = hi[hh]; }
  }

  v8f oa[4] = {};                            // O accumulators: 16 x 64 f32
  float m_i[8], l_i[8];
#pragma unroll
  for (int r = 0; r < 8; ++r) { m_i[r] = -1.0e30f; l_i[r] = 0.0f; }

  for (int kv0 = 0; kv0 < SK_; kv0 += 32) {
    if (kv0 + 32 < SK_)
      __builtin_prefetch(Kb + (size_t)(kv0 + 32 + n16) * DQK, 0, 1);

    // ---- S = Q * K^T (16 x 32), two 16-col subtiles -------------------------
    v8f s0 = {}, s1 = {};
#pragma unroll
    for (int kc = 0; kc < 2; ++kc) {
      v16h kf0 = *(const v16h*)(Kb + (size_t)(kv0 + n16) * DQK + kc * 32 + bkoff);
      s0 = wmma_f16(qf[kc], kf0, s0);
      v16h kf1 = *(const v16h*)(Kb + (size_t)(kv0 + 16 + n16) * DQK + kc * 32 + bkoff);
      s1 = wmma_f16(qf[kc], kf1, s1);
    }
    const bool mv0 = mb[kv0 + n16] != 0;
    const bool mv1 = mb[kv0 + 16 + n16] != 0;

    // ---- online softmax (rows live in VGPR slots, cols across 16 lanes) ----
    float mx[8];
#pragma unroll
    for (int r = 0; r < 8; ++r) {
      float a = mv0 ? s0[r] * scale : -1.0e30f;
      float c = mv1 ? s1[r] * scale : -1.0e30f;
      s0[r] = a; s1[r] = c;
      mx[r] = fmaxf(a, c);
    }
#pragma unroll
    for (int off = 1; off < 16; off <<= 1)
#pragma unroll
      for (int r = 0; r < 8; ++r)
        mx[r] = fmaxf(mx[r], __shfl_xor(mx[r], off, 32));

    float alpha[8], rs[8];
#pragma unroll
    for (int r = 0; r < 8; ++r) {
      float mnew = fmaxf(m_i[r], mx[r]);
      alpha[r] = __expf(m_i[r] - mnew);
      m_i[r] = mnew;
      float p0 = mv0 ? __expf(s0[r] - mnew) : 0.0f;
      float p1 = mv1 ? __expf(s1[r] - mnew) : 0.0f;
      s0[r] = p0; s1[r] = p1;
      rs[r] = p0 + p1;
    }
#pragma unroll
    for (int off = 1; off < 16; off <<= 1)
#pragma unroll
      for (int r = 0; r < 8; ++r)
        rs[r] += __shfl_xor(rs[r], off, 32);
#pragma unroll
    for (int r = 0; r < 8; ++r) l_i[r] = l_i[r] * alpha[r] + rs[r];

    // rescale O, stage P in LDS (C layout -> row-major 16x32)
#pragma unroll
    for (int r = 0; r < 8; ++r) {
      int m = rowoff + r;
      pw[m * 40 + n16]      = (_Float16)s0[r];
      pw[m * 40 + 16 + n16] = (_Float16)s1[r];
    }
#pragma unroll
    for (int j = 0; j < 4; ++j)
#pragma unroll
      for (int r = 0; r < 8; ++r) oa[j][r] *= alpha[r];

    // re-load P as A fragment (intra-wave LDS is in-order; compiler waits dscnt)
    v8h plo = *(const v8h*)&pw[arow * 40 + akoff];
    v8h phi = *(const v8h*)&pw[arow * 40 + akoff + 8];
    v16h pf;
#pragma unroll
    for (int hh = 0; hh < 8; ++hh) { pf[hh] = plo[hh]; pf[hh + 8] = phi[hh]; }

    // ---- O += P * V (V transposed -> contiguous B fragments) ---------------
#pragma unroll
    for (int j = 0; j < 4; ++j) {
      v16h vf = *(const v16h*)(Vb + (size_t)(j * 16 + n16) * SK_ + kv0 + bkoff);
      oa[j] = wmma_f16(pf, vf, oa[j]);
    }
  }

  // ---- epilogue: O / l, write [B,SQ,H*64] ----------------------------------
  float inv[8];
#pragma unroll
  for (int r = 0; r < 8; ++r) inv[r] = (l_i[r] > 0.0f) ? (1.0f / l_i[r]) : 0.0f;
  float* outp = out + ((size_t)b * SQ_ + q0 + rowoff) * NOUT + h * 64 + n16;
#pragma unroll
  for (int j = 0; j < 4; ++j)
#pragma unroll
    for (int r = 0; r < 8; ++r)
      outp[(size_t)r * NOUT + j * 16] = oa[j][r] * inv[r];
}

// ---------------------------------------------------------------------------
extern "C" void kernel_launch(void* const* d_in, const int* in_sizes, int n_in,
                              void* d_out, int out_size, void* d_ws, size_t ws_size,
                              hipStream_t stream) {
  const float* pre_qs = (const float*)d_in[0];
  const float* pre_ks = (const float*)d_in[1];
  const float* pre_vs = (const float*)d_in[2];
  const unsigned char* k_mask = (const unsigned char*)d_in[3];  // jnp bool -> 1B
  const float* q_w = (const float*)d_in[4];
  const float* q_b = (const float*)d_in[5];
  const float* k_w = (const float*)d_in[6];
  const float* k_b = (const float*)d_in[7];
  const float* v_w = (const float*)d_in[8];
  const float* v_b = (const float*)d_in[9];
  float* out = (float*)d_out;

  const size_t elems = (size_t)B_ * H_ * SQ_ * DQK;   // 8,388,608 halves each
  _Float16* qh = (_Float16*)d_ws;                     // [B,H,SQ,64]
  _Float16* kh = qh + elems;                          // [B,H,SK,64]
  _Float16* vt = kh + elems;                          // [B,H,64,SK]

  dim3 pgrid((B_ * SQ_) / 64, NOUT / 128, 1);         // (128, 8)
  proj16<0><<<pgrid, 256, 0, stream>>>(pre_qs, q_w, q_b, qh);
  proj16<0><<<pgrid, 256, 0, stream>>>(pre_ks, k_w, k_b, kh);
  proj16<1><<<pgrid, 256, 0, stream>>>(pre_vs, v_w, v_b, vt);

  dim3 agrid(SQ_ / 64, H_, B_);                       // (32, 16, 4)
  flash_attn<<<agrid, 128, 0, stream>>>(qh, kh, vt, k_mask, out);
}
